// UntiedHeadAssociativeLM_34368328303127
// MI455X (gfx1250) — compile-verified, matrix operand order checked by reading, not run on hardware
//
#include <hip/hip_runtime.h>

// ---------------------------------------------------------------------------
// UntiedHeadAssociativeLM for MI455X (gfx1250, wave32, WMMA + async-to-LDS)
// Pipeline:
//   1. f32->bf16 weight conversion (w_ih, w_hh, wq, wk, wh, wlm)
//   2. embedding gather -> x_bf16
//   3. WMMA GEMM: xW = x @ w_ih^T + b_ih              (2048 x 1536 x 256)
//   4. GRU scan, single WG, WMMA per step             (16 x 1536 x 512 per t)
//   5. WMMA GEMM: q,k = states @ wq/wk^T + b          (2048 x 128 x 512)
//   6. WMMA GEMM: scores = q @ k^T * 1/sqrt(M)        (1024 x 1024 x 128, x2)
//   7. causal softmax + sigmoid gate * memory_scale   (in-place on scores)
//   8. WMMA GEMM: hid = states @ wh^T + bh            (2048 x 256 x 512)
//   9. WMMA GEMM: out = hid @ wlm^T + blm -> d_out    (2048 x 32000 x 256)
//  10. scatter: out[b,i,ids[b,j]] += gated[b,i,j]     (atomic f32)
// GEMM uses GLOBAL_LOAD_ASYNC_TO_LDS_B128 double-buffering (ASYNCcnt).
// ---------------------------------------------------------------------------

typedef __attribute__((ext_vector_type(16))) __bf16 v16bf;
typedef __attribute__((ext_vector_type(8)))  float  v8f;

union BF16Frag { unsigned int u[8]; v16bf v; };

__device__ __forceinline__ unsigned short f32_to_bf16(float f) {
  union { float f; unsigned int u; } x; x.f = f;
  unsigned int r = x.u + 0x7FFFu + ((x.u >> 16) & 1u);   // round-to-nearest-even
  return (unsigned short)(r >> 16);
}

__device__ __forceinline__ float sigmoidf_fast(float x) {
  return 1.0f / (1.0f + __expf(-x));
}

// ---- CDNA5 async global->LDS copy (ASYNCcnt path, CDNA5 ISA) -----------

__device__ __forceinline__ void async_load_b128(unsigned lds_addr, const void* gaddr) {
  asm volatile("global_load_async_to_lds_b128 %0, %1, off"
               :: "v"(lds_addr), "v"(gaddr) : "memory");
}

__device__ __forceinline__ void async_wait_le3() {
#if __has_builtin(__builtin_amdgcn_s_wait_asynccnt)
  __builtin_amdgcn_s_wait_asynccnt(3);
#else
  asm volatile("s_wait_asynccnt 0x3" ::: "memory");
#endif
}

__device__ __forceinline__ void async_wait_le0() {
#if __has_builtin(__builtin_amdgcn_s_wait_asynccnt)
  __builtin_amdgcn_s_wait_asynccnt(0);
#else
  asm volatile("s_wait_asynccnt 0x0" ::: "memory");
#endif
}

// ----------------------------- elementwise helpers -------------------------

__global__ void cvt_bf16_kernel(const float* __restrict__ src,
                                unsigned short* __restrict__ dst, int n) {
  int i = blockIdx.x * blockDim.x + threadIdx.x;
  if (i < n) dst[i] = f32_to_bf16(src[i]);
}

// one block per token (2048), 256 threads = E
__global__ void embed_gather_kernel(const int* __restrict__ ids,
                                    const float* __restrict__ emb,
                                    unsigned short* __restrict__ xbf) {
  int tok = blockIdx.x;
  int e   = threadIdx.x;
  int id  = ids[tok];
  xbf[(size_t)tok * 256 + e] = f32_to_bf16(emb[(size_t)id * 256 + e]);
}

// ----------------------------- generic WMMA GEMM ---------------------------
// C[m][n] = alpha * sum_k A[m][k]*W[n][k] + bias[n]
// A: MxK bf16 row-major, W: NxK bf16 row-major (i.e. computes A @ W^T).
// M multiple of 128, N multiple of 64, K multiple of 32.
// Block = 256 threads = 8 waves; block tile 128x64; each wave owns a 32x32
// patch -> 4 WMMAs per BK=32 step from 2 A-frags + 2 B-frags.
// Tiles staged with async global->LDS b128, double-buffered on ASYNCcnt.

#define GBM 128
#define GBN 64
#define GBK 32
#define GLD 40   // padded LDS row stride (ushorts), keeps uint alignment

__global__ __launch_bounds__(256)
void gemm_bf16_wmma(const unsigned short* __restrict__ A,
                    const unsigned short* __restrict__ W,
                    const float* __restrict__ bias,
                    float* __restrict__ C,
                    unsigned short* __restrict__ Cbf,
                    int M, int N, int K, float alpha) {
  __shared__ unsigned short As[2][GBM * GLD];   // 2 x 10240 B
  __shared__ unsigned short Ws[2][GBN * GLD];   // 2 x  5120 B

  const int tid  = threadIdx.x;
  const int lane = tid & 31;
  const int wave = tid >> 5;
  const int m0   = blockIdx.y * GBM;
  const int n0   = blockIdx.x * GBN;
  const int wm   = wave & 3;     // 32-row group 0..3
  const int wn   = wave >> 2;    // 32-col group 0..1
  const int mlo  = lane & 15;
  const int kh   = lane >> 4;

  // async fill assignments
  const int ar = tid >> 1;          // 0..127, A row
  const int ac = (tid & 1) * 16;    // 0 / 16  (two b128 per thread)
  const int wr = tid >> 2;          // 0..63,  W row
  const int wc = (tid & 3) * 8;     // one b128 per thread

  const unsigned aL0 = (unsigned)(uintptr_t)&As[0][ar * GLD + ac];
  const unsigned aL1 = (unsigned)(uintptr_t)&As[1][ar * GLD + ac];
  const unsigned wL0 = (unsigned)(uintptr_t)&Ws[0][wr * GLD + wc];
  const unsigned wL1 = (unsigned)(uintptr_t)&Ws[1][wr * GLD + wc];
  const unsigned short* aG = A + (size_t)(m0 + ar) * K + ac;
  const unsigned short* wG = W + (size_t)(n0 + wr) * K + wc;

  v8f acc00 = {}, acc01 = {}, acc10 = {}, acc11 = {};
  const int KT = K / GBK;

  // prime buffer 0
  async_load_b128(aL0,      aG);
  async_load_b128(aL0 + 16, aG + 8);
  async_load_b128(wL0,      wG);

  for (int kt = 0; kt < KT; ++kt) {
    const int cur = kt & 1;
    if (kt + 1 < KT) {               // issue next tile into other buffer
      const unsigned short* aN = aG + (size_t)(kt + 1) * GBK;
      const unsigned short* wN = wG + (size_t)(kt + 1) * GBK;
      const unsigned aLn = cur ? aL0 : aL1;
      const unsigned wLn = cur ? wL0 : wL1;
      async_load_b128(aLn,      aN);
      async_load_b128(aLn + 16, aN + 8);
      async_load_b128(wLn,      wN);
      async_wait_le3();              // retire the 3 older (in-order) loads
    } else {
      async_wait_le0();
    }
    __syncthreads();

    BF16Frag a0, a1, b0, b1;
#pragma unroll
    for (int v = 0; v < 8; ++v) {
      // CDNA5 16-bit A/B frag layout: lanes 0-15 K in {0..7,16..23},
      // lanes 16-31 K in {8..15,24..31}; 2 packed bf16 per VGPR.
      const int k = (v < 4) ? (2 * v + 8 * kh) : (16 + 2 * (v - 4) + 8 * kh);
      a0.u[v] = *(const unsigned int*)&As[cur][(wm * 32 + mlo) * GLD + k];
      a1.u[v] = *(const unsigned int*)&As[cur][(wm * 32 + 16 + mlo) * GLD + k];
      b0.u[v] = *(const unsigned int*)&Ws[cur][(wn * 32 + mlo) * GLD + k];
      b1.u[v] = *(const unsigned int*)&Ws[cur][(wn * 32 + 16 + mlo) * GLD + k];
    }
    acc00 = __builtin_amdgcn_wmma_f32_16x16x32_bf16(false, a0.v, false, b0.v, (short)0, acc00, false, false);
    acc01 = __builtin_amdgcn_wmma_f32_16x16x32_bf16(false, a0.v, false, b1.v, (short)0, acc01, false, false);
    acc10 = __builtin_amdgcn_wmma_f32_16x16x32_bf16(false, a1.v, false, b0.v, (short)0, acc10, false, false);
    acc11 = __builtin_amdgcn_wmma_f32_16x16x32_bf16(false, a1.v, false, b1.v, (short)0, acc11, false, false);
    __syncthreads();                 // readers done before buffer reuse
  }

  // C/D layout: VGPR v, lane l -> M = v + 8*(l>>4), N = l&15
#pragma unroll
  for (int v = 0; v < 8; ++v) {
    const int gm0 = m0 + wm * 32 + v + 8 * kh;
    const int gm1 = gm0 + 16;
    const int gn0 = n0 + wn * 32 + mlo;
    const int gn1 = gn0 + 16;
    const float bn0 = bias ? bias[gn0] : 0.0f;
    const float bn1 = bias ? bias[gn1] : 0.0f;
    const float v00 = alpha * acc00[v] + bn0;
    const float v01 = alpha * acc01[v] + bn1;
    const float v10 = alpha * acc10[v] + bn0;
    const float v11 = alpha * acc11[v] + bn1;
    if (C) {
      C[(size_t)gm0 * N + gn0] = v00;
      C[(size_t)gm0 * N + gn1] = v01;
      C[(size_t)gm1 * N + gn0] = v10;
      C[(size_t)gm1 * N + gn1] = v11;
    }
    if (Cbf) {
      Cbf[(size_t)gm0 * N + gn0] = f32_to_bf16(v00);
      Cbf[(size_t)gm0 * N + gn1] = f32_to_bf16(v01);
      Cbf[(size_t)gm1 * N + gn0] = f32_to_bf16(v10);
      Cbf[(size_t)gm1 * N + gn1] = f32_to_bf16(v11);
    }
  }
}

// ----------------------------- GRU scan ------------------------------------
// Single workgroup, 1024 threads = 32 waves.  Per step:
//   gh(16x1536) = h(16x512, batch padded to 16) @ w_hh^T  via WMMA bf16
//   (each wave owns 3 N-tiles of 16 -> 96 tiles, 48 WMMAs/wave/step)
// then f32 gate math, h kept in f32 LDS, bf16 mirror feeds next step's WMMA.

__global__ __launch_bounds__(1024)
void gru_scan_kernel(const float* __restrict__ xW,            // (2,1024,1536) f32
                     const unsigned short* __restrict__ whh,  // (1536,512) bf16
                     const float* __restrict__ bhh,           // (1536)
                     float* __restrict__ states,              // (2,1024,512) f32
                     unsigned short* __restrict__ states_bf)  // same, bf16
{
  __shared__ unsigned short hbf[16 * 512];  // A-matrix, rows 2..15 stay zero
  __shared__ float ghb[2 * 1536];
  __shared__ float hcur[2 * 512];

  const int tid  = threadIdx.x;
  const int lane = tid & 31;
  const int wave = tid >> 5;
  const int m    = lane & 15;
  const int kh   = lane >> 4;

  for (int i = tid; i < 16 * 512; i += 1024) hbf[i] = 0;
  hcur[tid] = 0.0f;
  __syncthreads();

  for (int t = 0; t < 1024; ++t) {
    v8f c0 = {}, c1 = {}, c2 = {};
#pragma unroll 4
    for (int kc = 0; kc < 16; ++kc) {
      const int kb = kc * 32;
      BF16Frag a;
#pragma unroll
      for (int v = 0; v < 8; ++v) {
        const int k = kb + ((v < 4) ? (2 * v + 8 * kh) : (16 + 2 * (v - 4) + 8 * kh));
        a.u[v] = *(const unsigned int*)&hbf[m * 512 + k];
      }
      BF16Frag bf0, bf1, bf2;
      const int nb = wave * 48 + m;
#pragma unroll
      for (int v = 0; v < 8; ++v) {
        const int k = kb + ((v < 4) ? (2 * v + 8 * kh) : (16 + 2 * (v - 4) + 8 * kh));
        bf0.u[v] = *(const unsigned int*)&whh[(size_t)(nb)      * 512 + k];
        bf1.u[v] = *(const unsigned int*)&whh[(size_t)(nb + 16) * 512 + k];
        bf2.u[v] = *(const unsigned int*)&whh[(size_t)(nb + 32) * 512 + k];
      }
      c0 = __builtin_amdgcn_wmma_f32_16x16x32_bf16(false, a.v, false, bf0.v, (short)0, c0, false, false);
      c1 = __builtin_amdgcn_wmma_f32_16x16x32_bf16(false, a.v, false, bf1.v, (short)0, c1, false, false);
      c2 = __builtin_amdgcn_wmma_f32_16x16x32_bf16(false, a.v, false, bf2.v, (short)0, c2, false, false);
    }
    if (lane < 16) {
      const int n = wave * 48 + lane;
      ghb[n]             = c0[0] + bhh[n];
      ghb[1536 + n]      = c0[1] + bhh[n];
      ghb[n + 16]        = c1[0] + bhh[n + 16];
      ghb[1536 + n + 16] = c1[1] + bhh[n + 16];
      ghb[n + 32]        = c2[0] + bhh[n + 32];
      ghb[1536 + n + 32] = c2[1] + bhh[n + 32];
    }
    __syncthreads();
    {
      const int b = tid >> 9;     // batch
      const int j = tid & 511;    // hidden index
      const float* xw = xW + ((size_t)(b * 1024 + t)) * 1536;
      const float xr = xw[j], xz = xw[512 + j], xn = xw[1024 + j];
      const float hr = ghb[b * 1536 + j];
      const float hz = ghb[b * 1536 + 512 + j];
      const float hn = ghb[b * 1536 + 1024 + j];
      const float r  = sigmoidf_fast(xr + hr);
      const float z  = sigmoidf_fast(xz + hz);
      const float nn = tanhf(xn + r * hn);
      const float h  = (1.0f - z) * nn + z * hcur[tid];
      hcur[tid] = h;
      hbf[b * 512 + j] = f32_to_bf16(h);
      const size_t so = ((size_t)(b * 1024 + t)) * 512 + j;
      states[so]    = h;
      states_bf[so] = f32_to_bf16(h);
    }
    __syncthreads();
  }
}

// ----------------------- causal softmax + gate -----------------------------
// one block (256 thr) per row (b,i); in-place scores -> gated attn weights.
// gated[i,j] = softmax_{j<i}(scores[i,:])[j] * sigmoid(states[b,i]·wg + bg) * ms
// row i==0 -> all zeros (matches reference renormalization semantics).

__global__ __launch_bounds__(256)
void softmax_gate_kernel(float* __restrict__ sc,
                         const float* __restrict__ states,
                         const float* __restrict__ wg,
                         const float* __restrict__ bg,
                         const float* __restrict__ mscale) {
  const int row = blockIdx.x;          // 0..2047 = b*1024 + i
  const int i   = row & 1023;
  const int tid = threadIdx.x;
  float* s = sc + (size_t)row * 1024;
  __shared__ float red[256];

  // gate = sigmoid(states[row]·wg + bg) * memory_scale
  const float* st = states + (size_t)row * 512;
  red[tid] = st[tid] * wg[tid] + st[tid + 256] * wg[tid + 256];
  __syncthreads();
  for (int o = 128; o > 0; o >>= 1) {
    if (tid < o) red[tid] += red[tid + o];
    __syncthreads();
  }
  const float gscale = sigmoidf_fast(red[0] + bg[0]) * mscale[0];
  __syncthreads();

  // max over valid j < i
  float mx = -3.4e38f;
  for (int j = tid; j < i; j += 256) mx = fmaxf(mx, s[j]);
  red[tid] = mx;
  __syncthreads();
  for (int o = 128; o > 0; o >>= 1) {
    if (tid < o) red[tid] = fmaxf(red[tid], red[tid + o]);
    __syncthreads();
  }
  mx = red[0];
  __syncthreads();

  float sm = 0.0f;
  for (int j = tid; j < i; j += 256) sm += __expf(s[j] - mx);
  red[tid] = sm;
  __syncthreads();
  for (int o = 128; o > 0; o >>= 1) {
    if (tid < o) red[tid] += red[tid + o];
    __syncthreads();
  }
  sm = red[0];

  const float inv = (i > 0) ? (gscale / fmaxf(sm, 1e-20f)) : 0.0f;
  for (int j = tid; j < 1024; j += 256)
    s[j] = (j < i) ? __expf(s[j] - mx) * inv : 0.0f;
}

// ----------------------------- scatter-add ---------------------------------
// out[b,i, ids[b,j]] += gated[b,i,j]

__global__ __launch_bounds__(256)
void scatter_kernel(float* __restrict__ out,
                    const float* __restrict__ gated,
                    const int* __restrict__ ids) {
  const int row = blockIdx.x;          // b*1024 + i
  const int b   = row >> 10;
  const float* g = gated + (size_t)row * 1024;
  float* o = out + (size_t)row * 32000;
  const int* id = ids + b * 1024;
  for (int j = threadIdx.x; j < 1024; j += 256) {
    const float v = g[j];
    if (v != 0.0f) atomicAdd(&o[id[j]], v);
  }
}

// ----------------------------- launcher ------------------------------------

extern "C" void kernel_launch(void* const* d_in, const int* in_sizes, int n_in,
                              void* d_out, int out_size, void* d_ws, size_t ws_size,
                              hipStream_t stream) {
  (void)in_sizes; (void)n_in; (void)out_size; (void)ws_size;
  constexpr int E = 256, H = 512, H3 = 1536, Mm = 128, V = 32000, L = 1024, B = 2;
  constexpr int BL = B * L;  // 2048 tokens

  const int*   ids  = (const int*)  d_in[0];
  const float* emb  = (const float*)d_in[1];
  const float* w_ih = (const float*)d_in[2];
  const float* w_hh = (const float*)d_in[3];
  const float* b_ih = (const float*)d_in[4];
  const float* b_hh = (const float*)d_in[5];
  const float* wq   = (const float*)d_in[6];
  const float* bq   = (const float*)d_in[7];
  const float* wk   = (const float*)d_in[8];
  const float* bk   = (const float*)d_in[9];
  const float* wg   = (const float*)d_in[10];
  const float* bg   = (const float*)d_in[11];
  const float* wh   = (const float*)d_in[12];
  const float* bh   = (const float*)d_in[13];
  const float* wlm  = (const float*)d_in[14];
  const float* blm  = (const float*)d_in[15];
  const float* ms   = (const float*)d_in[16];
  float* out = (float*)d_out;

  // workspace carve-out (256B aligned slices)
  char* p = (char*)d_ws;
  auto carve = [&](size_t bytes) {
    void* r = (void*)p;
    p += (bytes + 255) & ~((size_t)255);
    return r;
  };
  unsigned short* wih_bf = (unsigned short*)carve((size_t)H3 * E * 2);
  unsigned short* whh_bf = (unsigned short*)carve((size_t)H3 * H * 2);
  unsigned short* wq_bf  = (unsigned short*)carve((size_t)Mm * H * 2);
  unsigned short* wk_bf  = (unsigned short*)carve((size_t)Mm * H * 2);
  unsigned short* wh_bf  = (unsigned short*)carve((size_t)E * H * 2);
  unsigned short* wlm_bf = (unsigned short*)carve((size_t)V * E * 2);
  unsigned short* x_bf   = (unsigned short*)carve((size_t)BL * E * 2);
  float*          xW     = (float*)         carve((size_t)BL * H3 * 4);
  float*          states = (float*)         carve((size_t)BL * H * 4);
  unsigned short* st_bf  = (unsigned short*)carve((size_t)BL * H * 2);
  unsigned short* q_bf   = (unsigned short*)carve((size_t)BL * Mm * 2);
  unsigned short* k_bf   = (unsigned short*)carve((size_t)BL * Mm * 2);
  float*          scores = (float*)         carve((size_t)B * L * L * 4);
  unsigned short* hid_bf = (unsigned short*)carve((size_t)BL * E * 2);

  auto cvt = [&](const float* s, unsigned short* d, int n) {
    cvt_bf16_kernel<<<(n + 255) / 256, 256, 0, stream>>>(s, d, n);
  };
  auto gemm = [&](const unsigned short* A, const unsigned short* W,
                  const float* bias, float* C, unsigned short* Cbf,
                  int M, int N, int K, float alpha) {
    dim3 grid(N / GBN, M / GBM);
    gemm_bf16_wmma<<<grid, 256, 0, stream>>>(A, W, bias, C, Cbf, M, N, K, alpha);
  };

  // 1. weight conversion
  cvt(w_ih, wih_bf, H3 * E);
  cvt(w_hh, whh_bf, H3 * H);
  cvt(wq,   wq_bf,  Mm * H);
  cvt(wk,   wk_bf,  Mm * H);
  cvt(wh,   wh_bf,  E * H);
  cvt(wlm,  wlm_bf, V * E);

  // 2. embedding gather
  embed_gather_kernel<<<BL, E, 0, stream>>>(ids, emb, x_bf);

  // 3. xW = x @ w_ih^T + b_ih
  gemm(x_bf, wih_bf, b_ih, xW, nullptr, BL, H3, E, 1.0f);

  // 4. GRU scan (serial over L, one workgroup, WMMA inner matmul)
  gru_scan_kernel<<<1, 1024, 0, stream>>>(xW, whh_bf, b_hh, states, st_bf);

  // 5. q, k projections (bf16 outputs feed the scores GEMM)
  gemm(st_bf, wq_bf, bq, nullptr, q_bf, BL, Mm, H, 1.0f);
  gemm(st_bf, wk_bf, bk, nullptr, k_bf, BL, Mm, H, 1.0f);

  // 6. scores[b] = q[b] @ k[b]^T / sqrt(M)
  const float rs = 0.08838834764831845f;  // 1/sqrt(128)
  for (int b = 0; b < B; ++b)
    gemm(q_bf + (size_t)b * L * Mm, k_bf + (size_t)b * L * Mm, nullptr,
         scores + (size_t)b * L * L, nullptr, L, L, Mm, rs);

  // 7. causal softmax * gate * memory_scale (in-place -> gated)
  softmax_gate_kernel<<<BL, 256, 0, stream>>>(scores, states, wg, bg, ms);

  // 8. hid = states @ wh^T + bh (bf16 out)
  gemm(st_bf, wh_bf, bh, nullptr, hid_bf, BL, E, H, 1.0f);

  // 9. base logits -> d_out (writes every element; no pre-zero needed)
  gemm(hid_bf, wlm_bf, blm, out, nullptr, BL, V, E, 1.0f);

  // 10. scatter-add associative memory
  scatter_kernel<<<BL, 256, 0, stream>>>(out, scores, ids);
}
